// MultiHeadAttention_67010079752298
// MI455X (gfx1250) — matrix operand order repaired and checked by reading, compiled, not run on hardware
//
#include <hip/hip_runtime.h>
#include <hip/hip_bf16.h>

#define BB 2
#define TT 2048
#define DD 1024
#define HH 16
#define DH 64

typedef __attribute__((ext_vector_type(16))) __bf16 v16bf;
typedef __attribute__((ext_vector_type(8)))  float  v8f;
typedef __attribute__((ext_vector_type(4)))  unsigned int u32x4;
typedef int v4i_vs __attribute__((vector_size(16)));   // matches builtin param type

#if defined(__gfx1250__) && __has_builtin(__builtin_amdgcn_global_load_async_to_lds_b128) && __has_builtin(__builtin_amdgcn_s_wait_asynccnt)
#define HAVE_ASYNC 1
#else
#define HAVE_ASYNC 0
#endif

union Frag {
    v16bf v;
    u32x4 q[2];
    unsigned int w[8];
    unsigned short s[16];
};

__device__ __forceinline__ unsigned short f2bf(float f) {
    unsigned int u = __float_as_uint(f);
    u += 0x7FFFu + ((u >> 16) & 1u);   // round-to-nearest-even
    return (unsigned short)(u >> 16);
}

__device__ __forceinline__ v8f wmma_bf16(v16bf a, v16bf b, v8f c) {
    return __builtin_amdgcn_wmma_f32_16x16x32_bf16(false, a, false, b, (short)0, c, false, false);
}

// 16-byte global -> LDS copy; async (ASYNCcnt) when the gfx1250 builtin exists.
__device__ __forceinline__ void async_copy16(void* lds_dst, const void* gsrc) {
#if HAVE_ASYNC
    __builtin_amdgcn_global_load_async_to_lds_b128(
        (__attribute__((address_space(1))) v4i_vs*)gsrc,
        (__attribute__((address_space(3))) v4i_vs*)lds_dst, 0, 0);
#else
    *(u32x4*)lds_dst = *(const u32x4*)gsrc;
#endif
}

__device__ __forceinline__ void wait_async() {
#if HAVE_ASYNC
    __builtin_amdgcn_s_wait_asynccnt(0);
#endif
}

// Packed fragment in global memory: 32 lanes x 32 bytes = 1KB per 16x32 (A) / 32x16 (B) tile.
__device__ __forceinline__ v16bf load_frag_g(const u32x4* tile, int lane) {
    Frag f;
    f.q[0] = tile[lane * 2 + 0];
    f.q[1] = tile[lane * 2 + 1];
    return f.v;
}
__device__ __forceinline__ void store_frag_g(u32x4* tile, int lane, v16bf vv) {
    Frag f; f.v = vv;
    tile[lane * 2 + 0] = f.q[0];
    tile[lane * 2 + 1] = f.q[1];
}
// Packed fragment staged in LDS (1KB tile): two ds_load_b128 per lane.
__device__ __forceinline__ v16bf load_frag_lds_packed(const unsigned char* tile, int lane) {
    Frag f;
    f.q[0] = *(const u32x4*)(tile + lane * 32);
    f.q[1] = *(const u32x4*)(tile + lane * 32 + 16);
    return f.v;
}

// A-fragment (16x32 bf16) gather from row-major bf16-bits LDS; contiguous-K layout:
// lane holds row r=lane&15; VGPR j packs K pair k0 = 16*hh+2j.
__device__ __forceinline__ v16bf load_afrag_lds(const unsigned short* base, int stride, int lane) {
    int r = lane & 15, hh = lane >> 4;
    const unsigned short* row = base + r * stride;
    Frag f;
#pragma unroll
    for (int j = 0; j < 8; ++j) {
        int k0 = 16 * hh + 2 * j;
        f.w[j] = *(const unsigned int*)(row + k0);
    }
    return f.v;
}
// Same gather from f32 LDS with on-the-fly bf16 conversion (pair loads -> ds_load_b64).
__device__ __forceinline__ v16bf load_afrag_lds_f32(const float* base, int stride, int lane) {
    int r = lane & 15, hh = lane >> 4;
    const float* row = base + r * stride;
    Frag f;
#pragma unroll
    for (int j = 0; j < 8; ++j) {
        int k0 = 16 * hh + 2 * j;
        float lo = row[k0], hi = row[k0 + 1];
        f.w[j] = (unsigned int)f2bf(lo) | ((unsigned int)f2bf(hi) << 16);
    }
    return f.v;
}

// B-fragment where the source is [n][k] row-major (contraction contiguous), e.g. K rows=keys.
__device__ __forceinline__ v16bf load_btfrag_lds(const unsigned short* base, int stride, int lane) {
    int n = lane & 15, hh = lane >> 4;
    const unsigned int* row = (const unsigned int*)(base + n * stride);
    Frag f;
#pragma unroll
    for (int j = 0; j < 8; ++j) f.w[j] = row[8 * hh + j];
    return f.v;
}

// B-fragment where the source is [k][n] row-major (contraction strided), e.g. V, W.
__device__ __forceinline__ v16bf load_bfrag_lds(const unsigned short* base, int stride, int lane) {
    int n = lane & 15, hh = lane >> 4;
    Frag f;
#pragma unroll
    for (int j = 0; j < 8; ++j) {
        int k = 16 * hh + 2 * j;
        f.s[2 * j]     = base[(size_t)k * stride + n];
        f.s[2 * j + 1] = base[(size_t)(k + 1) * stride + n];
    }
    return f.v;
}

// ---------------------------------------------------------------------------
// Kernel 1: convert f32 weight matrix [K][N] row-major -> bf16 packed B-fragments.
__global__ void pack_b_frags(const float* __restrict__ W, unsigned int* __restrict__ dst,
                             int ntiles, int ldw, int nwords) {
    int t = blockIdx.x * blockDim.x + threadIdx.x;
    if (t >= nwords) return;
    int j = t & 7;
    int lane = (t >> 3) & 31;
    int tile = t >> 8;
    int nt = tile % ntiles;
    int kc = tile / ntiles;
    int hh = lane >> 4;
    int n = 16 * nt + (lane & 15);
    int k = 32 * kc + 16 * hh + 2 * j;
    unsigned int lo = f2bf(W[(size_t)k * ldw + n]);
    unsigned int hi = f2bf(W[(size_t)(k + 1) * ldw + n]);
    dst[t] = lo | (hi << 16);
}

// ---------------------------------------------------------------------------
// Kernel 2: per-head QKV projection + repack into WMMA fragment layouts.
// Block = 256 thr (8 waves) handles 128 rows of one (b,h). grid = B*H*(T/128) = 512.
__launch_bounds__(256)
__global__ void qkv_pack_kernel(const float* __restrict__ x,
                                const u32x4* __restrict__ wqp, const u32x4* __restrict__ wkp,
                                const u32x4* __restrict__ wvp,
                                const float* __restrict__ bq, const float* __restrict__ bk,
                                const float* __restrict__ bv,
                                u32x4* __restrict__ qp, u32x4* __restrict__ kp,
                                u32x4* __restrict__ vp) {
    __shared__ __align__(16) float          xf[128 * 64];   // 32 KB, raw f32 slab
    __shared__ __align__(16) unsigned short qs[128 * 64];
    __shared__ __align__(16) unsigned short ks[128 * 64];
    __shared__ __align__(16) unsigned short vs[128 * 64];

    const int tid  = threadIdx.x;
    const int lane = tid & 31;
    const int wave = tid >> 5;
    const int slab = blockIdx.x & 15;          // T/128 slabs
    const int bh   = blockIdx.x >> 4;          // 0..31
    const int h    = bh & 15;
    const int b    = bh >> 4;
    const int t0   = slab * 128;

    // Phase 1: async-copy x slab (head h channels, f32) straight into LDS
    {
        const float* gsrc = x + (size_t)(b * TT + t0) * DD + h * DH;
        for (int c = tid; c < 2048; c += 256) {     // 2048 x 16B = 32 KB
            int row = c >> 4, col4 = (c & 15) * 4;
            async_copy16(&xf[row * 64 + col4], gsrc + (size_t)row * DD + col4);
        }
        wait_async();
    }
    __syncthreads();

    // Phase 2: each wave computes its 16-row Q/K/V tiles (K-dim 64 = 2 WMMA chunks)
    {
        const int nn = lane & 15, hh = lane >> 4;
        v16bf a0 = load_afrag_lds_f32(xf + wave * 16 * 64 +  0, 64, lane);
        v16bf a1 = load_afrag_lds_f32(xf + wave * 16 * 64 + 32, 64, lane);
        const u32x4* wsrc[3] = { wqp, wkp, wvp };
        const float* bsrc[3] = { bq, bk, bv };
        unsigned short* osrc[3] = { qs, ks, vs };
#pragma unroll
        for (int m = 0; m < 3; ++m) {
#pragma unroll
            for (int nt = 0; nt < 4; ++nt) {
                float bias = bsrc[m][h * DH + 16 * nt + nn];
                v8f acc;
#pragma unroll
                for (int v = 0; v < 8; ++v) acc[v] = bias;
                acc = wmma_bf16(a0, load_frag_g(wsrc[m] + (size_t)((h * 2 + 0) * 4 + nt) * 64, lane), acc);
                acc = wmma_bf16(a1, load_frag_g(wsrc[m] + (size_t)((h * 2 + 1) * 4 + nt) * 64, lane), acc);
#pragma unroll
                for (int v = 0; v < 8; ++v)
                    osrc[m][(wave * 16 + v + 8 * hh) * 64 + 16 * nt + nn] = f2bf(acc[v]);
            }
        }
    }
    __syncthreads();

    // Phase 3: repack to global fragment layouts (48 tiles, 6 per wave)
    for (int task = wave; task < 48; task += 8) {
        if (task < 16) {                       // Q -> A-fragments
            int mtl = task >> 1, kc = task & 1;
            v16bf f = load_afrag_lds(qs + mtl * 16 * 64 + 32 * kc, 64, lane);
            int mt = slab * 8 + mtl;
            store_frag_g(qp + ((size_t)(bh * 128 + mt) * 2 + kc) * 64, lane, f);
        } else if (task < 32) {                // K -> B-fragments of K^T
            int idx = task - 16, nbl = idx >> 1, kc = idx & 1;
            v16bf f = load_btfrag_lds(ks + nbl * 16 * 64 + 32 * kc, 64, lane);
            int nb = slab * 8 + nbl;
            store_frag_g(kp + ((size_t)(bh * 128 + nb) * 2 + kc) * 64, lane, f);
        } else {                               // V -> B-fragments
            int idx = task - 32, kbl = idx >> 2, nt = idx & 3;
            v16bf f = load_bfrag_lds(vs + kbl * 32 * 64 + 16 * nt, 64, lane);
            int kb = slab * 4 + kbl;
            store_frag_g(vp + ((size_t)(bh * 64 + kb) * 4 + nt) * 64, lane, f);
        }
    }
}

// ---------------------------------------------------------------------------
// Kernel 3: flash attention. One wave per (b,h,16-query tile); 8 waves/block; grid=512.
__launch_bounds__(256)
__global__ void attn_kernel(const u32x4* __restrict__ qp, const u32x4* __restrict__ kp,
                            const u32x4* __restrict__ vp, u32x4* __restrict__ op) {
    __shared__ __align__(16) unsigned short ps[8][16 * 32];
    __shared__ __align__(16) unsigned short os[8][16 * 64];

    const int lane = threadIdx.x & 31;
    const int wave = threadIdx.x >> 5;
    const int gw = blockIdx.x * 8 + wave;      // 0..4095
    const int qt = gw & 127;
    const int bh = gw >> 7;
    const int h  = bh & 15;
    const int b  = bh >> 4;
    const int nn = lane & 15, hh = lane >> 4;
    const float scale = 0.125f;                // 1/sqrt(64)
    const float NEG = -3.0e38f;

    v16bf qf0 = load_frag_g(qp + ((size_t)(bh * 128 + qt) * 2 + 0) * 64, lane);
    v16bf qf1 = load_frag_g(qp + ((size_t)(bh * 128 + qt) * 2 + 1) * 64, lane);

    float mrow[8], lrow[8];
    v8f oacc[4];
#pragma unroll
    for (int v = 0; v < 8; ++v) { mrow[v] = NEG; lrow[v] = 0.f; }
#pragma unroll
    for (int nt = 0; nt < 4; ++nt)
#pragma unroll
        for (int v = 0; v < 8; ++v) oacc[nt][v] = 0.f;

    const int kbmax = (qt * 16 + 15) >> 5;     // causal bound, 32-key blocks
    for (int kb = 0; kb <= kbmax; ++kb) {
        if (kb < kbmax) {                      // global_prefetch_b8 next K/V tiles
            __builtin_prefetch(kp + ((size_t)(bh * 128 + 2 * (kb + 1)) * 2) * 64, 0, 1);
            __builtin_prefetch(vp + ((size_t)(bh * 64 + (kb + 1)) * 4) * 64, 0, 1);
        }
        float s0v[8], s1v[8], p0v[8], p1v[8], alpha[8];
#pragma unroll
        for (int si = 0; si < 2; ++si) {
            int nb = 2 * kb + si;
            v8f sa;
#pragma unroll
            for (int v = 0; v < 8; ++v) sa[v] = 0.f;
            sa = wmma_bf16(qf0, load_frag_g(kp + ((size_t)(bh * 128 + nb) * 2 + 0) * 64, lane), sa);
            sa = wmma_bf16(qf1, load_frag_g(kp + ((size_t)(bh * 128 + nb) * 2 + 1) * 64, lane), sa);
            int key = 32 * kb + 16 * si + nn;
#pragma unroll
            for (int v = 0; v < 8; ++v) {
                int qrow = qt * 16 + v + 8 * hh;
                float val = (key <= qrow) ? sa[v] * scale : NEG;
                if (si == 0) s0v[v] = val; else s1v[v] = val;
            }
        }
        // online softmax: row max/sum over 16-lane half-groups (C-layout rows v+8*hh)
#pragma unroll
        for (int v = 0; v < 8; ++v) {
            float mx = fmaxf(s0v[v], s1v[v]);
#pragma unroll
            for (int msk = 1; msk <= 8; msk <<= 1) mx = fmaxf(mx, __shfl_xor(mx, msk, 32));
            float mnew = fmaxf(mrow[v], mx);
            alpha[v] = __expf(mrow[v] - mnew);
            p0v[v] = __expf(s0v[v] - mnew);
            p1v[v] = __expf(s1v[v] - mnew);
            float psum = p0v[v] + p1v[v];
#pragma unroll
            for (int msk = 1; msk <= 8; msk <<= 1) psum += __shfl_xor(psum, msk, 32);
            lrow[v] = lrow[v] * alpha[v] + psum;
            mrow[v] = mnew;
        }
        // P (C layout) -> LDS -> A-fragment (DS ops are same-wave in-order)
#pragma unroll
        for (int v = 0; v < 8; ++v) {
            ps[wave][(v + 8 * hh) * 32 +      nn] = f2bf(p0v[v]);
            ps[wave][(v + 8 * hh) * 32 + 16 + nn] = f2bf(p1v[v]);
        }
        v16bf pf = load_afrag_lds(ps[wave], 32, lane);
#pragma unroll
        for (int nt = 0; nt < 4; ++nt) {
#pragma unroll
            for (int v = 0; v < 8; ++v) oacc[nt][v] *= alpha[v];
            oacc[nt] = wmma_bf16(pf, load_frag_g(vp + ((size_t)(bh * 64 + kb) * 4 + nt) * 64, lane), oacc[nt]);
        }
    }

    // epilogue: normalize, stage, emit A-fragments of concat-head activation
    float rinv[8];
#pragma unroll
    for (int v = 0; v < 8; ++v) rinv[v] = 1.0f / lrow[v];
#pragma unroll
    for (int nt = 0; nt < 4; ++nt)
#pragma unroll
        for (int v = 0; v < 8; ++v)
            os[wave][(v + 8 * hh) * 64 + 16 * nt + nn] = f2bf(oacc[nt][v] * rinv[v]);
    const int mt = b * 128 + qt;               // row tile in [B*T, D]
#pragma unroll
    for (int kc = 0; kc < 2; ++kc) {
        v16bf f = load_afrag_lds(os[wave] + 32 * kc, 64, lane);
        store_frag_g(op + ((size_t)mt * 32 + (2 * h + kc)) * 64, lane, f);
    }
}

// ---------------------------------------------------------------------------
// Kernel 4: output projection out = O[4096,1024] @ Wo[1024,1024] + bo (f32 out).
// Block: 8 waves -> 128x64 tile; Wo tiles double-buffered in LDS via async copies.
__launch_bounds__(256)
__global__ void proj_kernel(const u32x4* __restrict__ op, const unsigned char* __restrict__ wop,
                            const float* __restrict__ bo, float* __restrict__ out) {
    __shared__ __align__(16) unsigned char wbuf[2][4096];   // 4 B-tiles per kc step

    const int tid  = threadIdx.x;
    const int lane = tid & 31;
    const int wave = tid >> 5;
    const int bm = blockIdx.x >> 4;
    const int bn = blockIdx.x & 15;
    const int mt = bm * 8 + wave;              // 0..255
    const int nn = lane & 15, hh = lane >> 4;

    v8f acc[4];
#pragma unroll
    for (int nt = 0; nt < 4; ++nt) {
        float bias = bo[(bn * 4 + nt) * 16 + nn];
#pragma unroll
        for (int v = 0; v < 8; ++v) acc[nt][v] = bias;
    }

    // prologue: stage kc=0 weight chunk (4KB = 4 packed B-tiles, contiguous)
    async_copy16(&wbuf[0][tid * 16], wop + ((size_t)0 * 64 + bn * 4) * 1024 + tid * 16);
    wait_async();
    __syncthreads();

    for (int kc = 0; kc < 32; ++kc) {
        int cur = kc & 1, nxt = cur ^ 1;
        if (kc + 1 < 32)                       // issue async copy of next chunk
            async_copy16(&wbuf[nxt][tid * 16],
                         wop + ((size_t)(kc + 1) * 64 + bn * 4) * 1024 + tid * 16);

        v16bf a = load_frag_g(op + ((size_t)mt * 32 + kc) * 64, lane);
        if (kc < 31) __builtin_prefetch(op + ((size_t)mt * 32 + kc + 1) * 64, 0, 1);
#pragma unroll
        for (int nt = 0; nt < 4; ++nt)
            acc[nt] = wmma_bf16(a, load_frag_lds_packed(&wbuf[cur][nt * 1024], lane), acc[nt]);

        wait_async();                          // next chunk landed
        __syncthreads();                       // everyone done with cur before overwrite
    }
#pragma unroll
    for (int nt = 0; nt < 4; ++nt) {
        int ntg = bn * 4 + nt;
#pragma unroll
        for (int v = 0; v < 8; ++v)
            out[(size_t)(mt * 16 + v + 8 * hh) * DD + ntg * 16 + nn] = acc[nt][v];
    }
}

// ---------------------------------------------------------------------------
extern "C" void kernel_launch(void* const* d_in, const int* in_sizes, int n_in,
                              void* d_out, int out_size, void* d_ws, size_t ws_size,
                              hipStream_t stream) {
    (void)in_sizes; (void)n_in; (void)out_size; (void)ws_size;
    const float* x  = (const float*)d_in[0];
    const float* Wq = (const float*)d_in[1];
    const float* bq = (const float*)d_in[2];
    const float* Wk = (const float*)d_in[3];
    const float* bk = (const float*)d_in[4];
    const float* Wv = (const float*)d_in[5];
    const float* bv = (const float*)d_in[6];
    const float* Wo = (const float*)d_in[7];
    const float* bo = (const float*)d_in[8];
    float* out = (float*)d_out;

    char* ws = (char*)d_ws;
    size_t off = 0;
    unsigned int* wqp = (unsigned int*)(ws + off); off += 128u * 1024;      // H*2*4 tiles
    unsigned int* wkp = (unsigned int*)(ws + off); off += 128u * 1024;
    unsigned int* wvp = (unsigned int*)(ws + off); off += 128u * 1024;
    unsigned int* wop = (unsigned int*)(ws + off); off += 2048u * 1024;     // 32*64 tiles
    u32x4* qp = (u32x4*)(ws + off); off += 8u * 1024 * 1024;                // A-frags
    u32x4* kp = (u32x4*)(ws + off); off += 8u * 1024 * 1024;                // B-frags (K^T)
    u32x4* vp = (u32x4*)(ws + off); off += 8u * 1024 * 1024;                // B-frags (V)
    u32x4* op = (u32x4*)(ws + off); off += 8u * 1024 * 1024;                // A-frags (attn out)

    pack_b_frags<<<128,  256, 0, stream>>>(Wq, wqp, 4,  64,   32 * 4  * 256);
    pack_b_frags<<<128,  256, 0, stream>>>(Wk, wkp, 4,  64,   32 * 4  * 256);
    pack_b_frags<<<128,  256, 0, stream>>>(Wv, wvp, 4,  64,   32 * 4  * 256);
    pack_b_frags<<<2048, 256, 0, stream>>>(Wo, wop, 64, 1024, 32 * 64 * 256);
    qkv_pack_kernel<<<512, 256, 0, stream>>>(x, (const u32x4*)wqp, (const u32x4*)wkp,
                                             (const u32x4*)wvp, bq, bk, bv, qp, kp, vp);
    attn_kernel<<<512, 256, 0, stream>>>(qp, kp, vp, op);
    proj_kernel<<<512, 256, 0, stream>>>(op, (const unsigned char*)wop, bo, out);
}
